// StaticGQAAttention_27084063768614
// MI455X (gfx1250) — matrix operand order repaired and checked by reading, not compiled, not run on hardware
//
#include <hip/hip_runtime.h>
#include <hip/hip_bf16.h>
#include <cstdint>

typedef __bf16 bf16_t;
typedef __attribute__((ext_vector_type(16))) bf16_t v16bf;
typedef __attribute__((ext_vector_type(8)))  float  v8f;
typedef __attribute__((ext_vector_type(4)))  unsigned int u32x4;
typedef __attribute__((ext_vector_type(8)))  int  i32x8;
typedef __attribute__((ext_vector_type(4)))  int  i32x4;

#define N_HEADS 32
#define N_KV    8
#define HD      128
#define DIM     4096
#define KV_DIM  1024
#define MAX_SEQ 4096
#define POS     4095          // MAX_SEQ - T, T == 1 (static in reference)
#define SPLIT   16
#define CHUNK   (MAX_SEQ / SPLIT)   // 256 positions per block
#define SCALE   0.08838834764831843f // 1/sqrt(128)

// LDS K-tile layout: TDM pads 4 DWORDs after every 128 DWORDs (one 512B row),
// giving a row stride of 132 floats -> rows land 4 banks apart -> the 16-float
// per-lane B-fragment reads are bank-conflict-free.
#define KROW    132
#define KTILE_F (16 * KROW)   // 2112 floats per K tile in LDS

// workspace layout (float offsets)
#define Q_OFF    0            // 4096 : q after projection+rope
#define K_OFF    4096         // 1024 : new k (8 heads x 128), rope applied
#define V_OFF    5120         // 1024 : new v
#define ATT_OFF  6144         // 4096 : attention output (pre-wo)
#define PART_OFF 10240        // 512 partials * 520 floats
#define PART_STRIDE 520       // 4 m + 4 l + 4*128 acc

// ---------------------------------------------------------------------------
// TDM: issue an async DMA of one K tile (16 rows x 128 f32, row stride 128)
// from global memory into LDS with 4-DWORD row padding. D# built per the CDNA5
// ISA descriptor layout (group0: count/lds_addr/global_addr/type; group1:
// data_size=4B, pad_enable, pad_interval=128dw, pad_amount=4dw, dims/strides).
// Tracked with per-wave TENSORcnt. This toolchain's builtin takes 6 args:
// (g0, g1, g2, g3, g4, cpol) -- the trailing groups are zero/unused for 2D.
// ---------------------------------------------------------------------------
__device__ __forceinline__ void tdm_load_k_tile(const float* gsrc,
                                                unsigned lds_off) {
  unsigned long long ga = (unsigned long long)(uintptr_t)gsrc;
  u32x4 g0;
  g0[0] = 1u;                                        // count = 1 (valid D#)
  g0[1] = lds_off;                                   // lds_addr (bytes)
  g0[2] = (unsigned)(ga & 0xffffffffu);              // global_addr[31:0]
  g0[3] = (unsigned)((ga >> 32) & 0x1ffffffu)        // global_addr[56:32]
          | (2u << 30);                              // type = 2 (image)
  i32x8 g1;
  g1[0] = (int)((2u << 16)                           // data_size = 4 bytes
              | (1u << 20)                           // pad_enable
              | (6u << 22)                           // pad_interval = 128 dw
              | (3u << 25));                         // pad_amount  = 4 dw
  g1[1] = (int)(128u << 16);                         // tensor_dim0 = 128
  g1[2] = (int)(16u  << 16);                         // tensor_dim1 = 16
  g1[3] = (int)(128u << 16);                         // tile_dim0   = 128
  g1[4] = 16;                                        // tile_dim1   = 16
  g1[5] = 128;                                       // tensor_dim0_stride
  g1[6] = 0;
  g1[7] = 0;
  i32x4 gz  = {0, 0, 0, 0};                          // groups 2/3 unused (2D)
  i32x8 gz8 = {0, 0, 0, 0, 0, 0, 0, 0};              // trailing group unused
  __builtin_amdgcn_tensor_load_to_lds(g0, g1, gz, gz, gz8, 0);
}

// ---------------------------------------------------------------------------
// Kernel 1: fused QKV GEMV. rows 0..4095 -> q, 4096..5119 -> k, 5120..6143 -> v
// One wave32 per output row; x staged in LDS; float4 (b128) weight stream.
// ---------------------------------------------------------------------------
__global__ __launch_bounds__(256) void qkv_gemv_kernel(
    const float* __restrict__ x,  const float* __restrict__ wq,
    const float* __restrict__ wk, const float* __restrict__ wv,
    float* __restrict__ ws) {
  __shared__ float xs[DIM];
  for (int i = threadIdx.x; i < DIM / 4; i += 256)
    reinterpret_cast<float4*>(xs)[i] = reinterpret_cast<const float4*>(x)[i];
  __syncthreads();

  const int wave = threadIdx.x >> 5;
  const int lane = threadIdx.x & 31;
  const int row  = blockIdx.x * 8 + wave;          // 0..6143
  const float* wrow;
  if (row < DIM)               wrow = wq + (size_t)row * DIM;
  else if (row < DIM + KV_DIM) wrow = wk + (size_t)(row - DIM) * DIM;
  else                         wrow = wv + (size_t)(row - DIM - KV_DIM) * DIM;

  float s = 0.f;
  for (int j = lane * 4; j < DIM; j += 128) {
    __builtin_prefetch(wrow + j + 2048, 0, 0);     // global_prefetch ahead
    float4 w4 = *reinterpret_cast<const float4*>(wrow + j);
    float4 x4 = *reinterpret_cast<const float4*>(xs + j);
    s = fmaf(w4.x, x4.x, s); s = fmaf(w4.y, x4.y, s);
    s = fmaf(w4.z, x4.z, s); s = fmaf(w4.w, x4.w, s);
  }
  #pragma unroll
  for (int off = 16; off >= 1; off >>= 1) s += __shfl_xor(s, off);
  if (lane == 0) ws[row] = s;
}

// ---------------------------------------------------------------------------
// Kernel 2: RoPE on q (4096) and k (1024) in workspace, at position POS.
// ---------------------------------------------------------------------------
__global__ __launch_bounds__(256) void rope_kernel(
    const float* __restrict__ fc, const float* __restrict__ fs,
    float* __restrict__ ws) {
  const int idx = blockIdx.x * 256 + threadIdx.x;  // pair index
  if (idx >= (DIM + KV_DIM) / 2) return;
  const int e0 = idx * 2;
  const int i  = (e0 & (HD - 1)) >> 1;             // rotary pair within head
  const float c = fc[POS * (HD / 2) + i];
  const float s = fs[POS * (HD / 2) + i];
  const float x0 = ws[e0], x1 = ws[e0 + 1];
  ws[e0]     = x0 * c - x1 * s;
  ws[e0 + 1] = x0 * s + x1 * c;
}

// ---------------------------------------------------------------------------
// Kernel 3: flash-decode attention partials.
// grid = (SPLIT, N_KV); block = 128 threads = 4 waves; each wave owns 64
// positions. K tiles are double-buffered in LDS via TDM (per-wave TENSORcnt,
// no cross-wave sync). Scores: zero-padded Q_group(16x128,bf16) x K^T via
// v_wmma_f32_16x16x32_bf16. Online softmax per wave; P x V with lane-owned
// float4 of the 128-dim output (V loads are fully coalesced 512B rows).
// ---------------------------------------------------------------------------
__global__ __launch_bounds__(128) void attn_partial_kernel(
    const float* __restrict__ kbuf, const float* __restrict__ vbuf,
    float* __restrict__ ws) {
  extern __shared__ float smem[];                  // 4 waves * 2 * KTILE_F
  const int lane = threadIdx.x & 31;
  const int wave = threadIdx.x >> 5;
  const int h    = blockIdx.y;                     // kv head
  const int m    = lane & 15;
  const bool hig = lane >= 16;

  const float* q    = ws + Q_OFF;
  const float* knew = ws + K_OFF + h * HD;
  const float* vnew = ws + V_OFF + h * HD;

  float* kls = smem + wave * (2 * KTILE_F);        // this wave's two K buffers
  const unsigned kls_off =
      (unsigned)(uintptr_t)(void*)kls;             // LDS byte offset (addr[31:0])

  // A fragments: 16x32 bf16 layout per 32-wide K chunk (rows 4..15 zero).
  v16bf afrag[4];
  #pragma unroll
  for (int kc = 0; kc < 4; ++kc) {
    v16bf a = {};
    if (m < 4) {
      const float* qb = q + (h * 4 + m) * HD + kc * 32 + (hig ? 8 : 0);
      #pragma unroll
      for (int i = 0; i < 8; ++i) a[i]     = (bf16_t)qb[i];
      #pragma unroll
      for (int i = 0; i < 8; ++i) a[8 + i] = (bf16_t)qb[16 + i];
    }
    afrag[kc] = a;
  }

  float  m_run[4], l_run[4];
  float4 acc[4];
  #pragma unroll
  for (int j = 0; j < 4; ++j) {
    m_run[j] = -1e30f; l_run[j] = 0.f; acc[j] = make_float4(0.f, 0.f, 0.f, 0.f);
  }

  const int sbase = blockIdx.x * CHUNK + wave * 64;
  const float* kgbase = kbuf + (size_t)h * MAX_SEQ * HD;

  // Prologue: DMA tile 0 into buffer 0.
  tdm_load_k_tile(kgbase + (size_t)sbase * HD, kls_off);

  for (int t = 0; t < 4; ++t) {
    // Double buffer: issue tile t+1, then wait for tile t only.
    if (t < 3) {
      tdm_load_k_tile(kgbase + (size_t)(sbase + (t + 1) * 16) * HD,
                      kls_off + (unsigned)(((t + 1) & 1) * KTILE_F * 4));
      __builtin_amdgcn_s_wait_tensorcnt(1);
    } else {
      __builtin_amdgcn_s_wait_tensorcnt(0);
    }

    float* ks = kls + (t & 1) * KTILE_F;
    const int p0 = sbase + t * 16;

    // New-token K lives in workspace: patch the tile's last LDS row.
    if (p0 + 15 == POS)
      *reinterpret_cast<float4*>(ks + 15 * KROW + lane * 4) =
          *reinterpret_cast<const float4*>(knew + lane * 4);

    // B fragment from LDS: lane n<16 = position p0+n (K 0..7,16..23 of the 32-chunk),
    // lane 16+n = K 8..15,24..31. Padded KROW stride -> conflict-free.
    v8f c = {};
    #pragma unroll
    for (int kc = 0; kc < 4; ++kc) {
      const float* kb = ks + m * KROW + kc * 32 + (hig ? 8 : 0);
      v16bf b;
      #pragma unroll
      for (int i = 0; i < 8; ++i) { b[i] = (bf16_t)kb[i]; b[8 + i] = (bf16_t)kb[16 + i]; }
      c = __builtin_amdgcn_wmma_f32_16x16x32_bf16(
              false, afrag[kc], false, b, (short)0, c, false, false);
    }

    // Online softmax per q head (D[j][n] = c[j] on lane n<16).
    float pv[4];
    #pragma unroll
    for (int j = 0; j < 4; ++j) {
      float sc = (lane < 16) ? c[j] * SCALE : -1e30f;
      float tm = sc;
      tm = fmaxf(tm, __shfl_xor(tm, 1));
      tm = fmaxf(tm, __shfl_xor(tm, 2));
      tm = fmaxf(tm, __shfl_xor(tm, 4));
      tm = fmaxf(tm, __shfl_xor(tm, 8));
      tm = __shfl(tm, 0);
      float nm = fmaxf(m_run[j], tm);
      float pj = (lane < 16) ? __expf(sc - nm) : 0.f;
      float ts = pj;
      ts += __shfl_xor(ts, 1);
      ts += __shfl_xor(ts, 2);
      ts += __shfl_xor(ts, 4);
      ts += __shfl_xor(ts, 8);
      ts = __shfl(ts, 0);
      float corr = __expf(m_run[j] - nm);
      l_run[j] = l_run[j] * corr + ts;
      m_run[j] = nm;
      acc[j].x *= corr; acc[j].y *= corr; acc[j].z *= corr; acc[j].w *= corr;
      pv[j] = pj;
    }

    // P x V: lane owns dims [4*lane, 4*lane+4); broadcast P from owner lane.
    #pragma unroll 4
    for (int jj = 0; jj < 16; ++jj) {
      const int pp = p0 + jj;
      const float* vr = (pp == POS) ? vnew
                                    : (vbuf + ((size_t)h * MAX_SEQ + pp) * HD);
      float4 vv = *reinterpret_cast<const float4*>(vr + lane * 4);
      #pragma unroll
      for (int j = 0; j < 4; ++j) {
        float w = __shfl(pv[j], jj);
        acc[j].x = fmaf(w, vv.x, acc[j].x);
        acc[j].y = fmaf(w, vv.y, acc[j].y);
        acc[j].z = fmaf(w, vv.z, acc[j].z);
        acc[j].w = fmaf(w, vv.w, acc[j].w);
      }
    }
  }

  // Store partial (m, l uniform across lanes; acc distributed by lane).
  float* part = ws + PART_OFF +
                (size_t)((h * SPLIT + blockIdx.x) * 4 + wave) * PART_STRIDE;
  if (lane == 0) {
    #pragma unroll
    for (int j = 0; j < 4; ++j) { part[j] = m_run[j]; part[4 + j] = l_run[j]; }
  }
  #pragma unroll
  for (int j = 0; j < 4; ++j)
    *reinterpret_cast<float4*>(part + 8 + j * HD + lane * 4) = acc[j];
}

// ---------------------------------------------------------------------------
// Kernel 4: combine 64 partials per q head (log-sum-exp correction).
// ---------------------------------------------------------------------------
__global__ __launch_bounds__(128) void attn_combine_kernel(float* __restrict__ ws) {
  const int qh = blockIdx.x;                       // 0..31
  const int h  = qh >> 2, j = qh & 3;
  const int d  = threadIdx.x;                      // 0..127
  const float* pbase = ws + PART_OFF + (size_t)(h * SPLIT * 4) * PART_STRIDE;

  float M = -1e30f;
  for (int p = 0; p < SPLIT * 4; ++p)
    M = fmaxf(M, pbase[(size_t)p * PART_STRIDE + j]);
  float L = 0.f, o = 0.f;
  for (int p = 0; p < SPLIT * 4; ++p) {
    const float* pp = pbase + (size_t)p * PART_STRIDE;
    float w = __expf(pp[j] - M);
    L += w * pp[4 + j];
    o  = fmaf(w, pp[8 + j * HD + d], o);
  }
  ws[ATT_OFF + qh * HD + d] = o / L;
}

// ---------------------------------------------------------------------------
// Kernel 5: output projection GEMV over wo (64 MB stream).
// ---------------------------------------------------------------------------
__global__ __launch_bounds__(256) void out_gemv_kernel(
    const float* __restrict__ wo, const float* __restrict__ ws,
    float* __restrict__ out) {
  __shared__ float xs[DIM];
  const float* a = ws + ATT_OFF;
  for (int i = threadIdx.x; i < DIM / 4; i += 256)
    reinterpret_cast<float4*>(xs)[i] = reinterpret_cast<const float4*>(a)[i];
  __syncthreads();

  const int wave = threadIdx.x >> 5;
  const int lane = threadIdx.x & 31;
  const int row  = blockIdx.x * 8 + wave;
  const float* wrow = wo + (size_t)row * DIM;

  float s = 0.f;
  for (int j = lane * 4; j < DIM; j += 128) {
    __builtin_prefetch(wrow + j + 2048, 0, 0);
    float4 w4 = *reinterpret_cast<const float4*>(wrow + j);
    float4 x4 = *reinterpret_cast<const float4*>(xs + j);
    s = fmaf(w4.x, x4.x, s); s = fmaf(w4.y, x4.y, s);
    s = fmaf(w4.z, x4.z, s); s = fmaf(w4.w, x4.w, s);
  }
  #pragma unroll
  for (int off = 16; off >= 1; off >>= 1) s += __shfl_xor(s, off);
  if (lane == 0) out[row] = s;
}

// ---------------------------------------------------------------------------
extern "C" void kernel_launch(void* const* d_in, const int* in_sizes, int n_in,
                              void* d_out, int out_size, void* d_ws, size_t ws_size,
                              hipStream_t stream) {
  const float* x  = (const float*)d_in[0];
  const float* fc = (const float*)d_in[1];
  const float* fs = (const float*)d_in[2];
  const float* kb = (const float*)d_in[3];
  const float* vb = (const float*)d_in[4];
  const float* wq = (const float*)d_in[5];
  const float* wk = (const float*)d_in[6];
  const float* wv = (const float*)d_in[7];
  const float* wo = (const float*)d_in[8];
  (void)in_sizes; (void)n_in; (void)out_size; (void)ws_size;

  float* ws  = (float*)d_ws;
  float* out = (float*)d_out;

  qkv_gemv_kernel<<<768, 256, 0, stream>>>(x, wq, wk, wv, ws);
  rope_kernel<<<10, 256, 0, stream>>>(fc, fs, ws);
  const size_t attn_lds = (size_t)4 * 2 * KTILE_F * sizeof(float); // ~66 KB
  attn_partial_kernel<<<dim3(SPLIT, N_KV), 128, attn_lds, stream>>>(kb, vb, ws);
  attn_combine_kernel<<<32, 128, 0, stream>>>(ws);
  out_gemv_kernel<<<512, 256, 0, stream>>>(wo, ws, out);
}
